// PointTransformerLayer_29970281791909
// MI455X (gfx1250) — matrix operand order repaired and checked by reading, not compiled, not run on hardware
//
#include <hip/hip_runtime.h>
#include <hip/hip_bf16.h>

typedef __attribute__((ext_vector_type(16))) __bf16 v16bf;
typedef __attribute__((ext_vector_type(8)))  float  v8f;
typedef __attribute__((ext_vector_type(4)))  unsigned int u32x4;
typedef __attribute__((ext_vector_type(8)))  int i32x8;
typedef __attribute__((ext_vector_type(4)))  int i32x4;

#define B_SZ  4
#define N_PTS 4096
#define R_TOT (B_SZ * N_PTS)
#define KNN   16
#define CIN   64
#define COUT  128

static __device__ __forceinline__ v8f wmma_bf16(v16bf a, v16bf b, v8f c) {
  return __builtin_amdgcn_wmma_f32_16x16x32_bf16(false, a, false, b, (short)0, c,
                                                 false, false);
}

// B fragment (K x 16 tile) from an LDS weight stored column-major: w[n*Kdim + k].
// Lane L (<16): col n = 16t+L, K = 32q+{0..7,16..23} ; lane L (>=16): K = 32q+{8..15,24..31}.
static __device__ __forceinline__ v16bf load_b_frag(const __bf16* __restrict__ wlds,
                                                    int Kdim, int t, int q, int lane) {
  int n  = 16 * t + (lane & 15);
  int k0 = 32 * q + ((lane & 16) ? 8 : 0);
  const __bf16* p = wlds + n * Kdim + k0;
  v16bf b;
#pragma unroll
  for (int i = 0; i < 8; ++i) { b[i] = p[i]; b[8 + i] = p[16 + i]; }
  return b;
}

// ---------------------------------------------------------------------------
// TDM: 1-D DMA of `nbytes` (multiple of 8, <= 65535*8) from global to LDS.
// Descriptor per CDNA5 ISA: group0 {count=1, lds_addr,
// global_addr, type=2}, group1 {data_size=8B, tensor_dim0=tile_dim0=n8,
// dim1=1, strides=n8}. Groups 2/3 unused (<=2D). Completion via TENSORcnt.
// ---------------------------------------------------------------------------
static __device__ __forceinline__ void tdm_load_lds_1d(unsigned lds_off,
                                                       const void* gsrc,
                                                       unsigned nbytes) {
  unsigned long long ga = (unsigned long long)gsrc;
  unsigned n8 = nbytes >> 3;                            // 8-byte elements
  u32x4 g0;
  g0[0] = 1u;                                           // count=1 (valid)
  g0[1] = lds_off;                                      // lds_addr (bytes)
  g0[2] = (unsigned)ga;                                 // global_addr[31:0]
  g0[3] = (unsigned)((ga >> 32) & 0x01FFFFFFull) | (2u << 30); // addr[56:32]|type=2
  i32x8 g1;
  g1[0] = (int)(3u << 16);                              // data_size=8B, mask=0
  g1[1] = (int)((n8 & 0xFFFFu) << 16);                  // tensor_dim0[15:0]
  g1[2] = (int)((n8 >> 16) | (1u << 16));               // dim0[31:16] | tensor_dim1=1
  g1[3] = (int)(n8 << 16);                              // dim1 hi=0 | tile_dim0=n8
  g1[4] = 1;                                            // tile_dim1=1, tile_dim2=0
  g1[5] = (int)n8;                                      // tensor_dim0_stride lo
  g1[6] = (int)((n8 & 0xFFFFu) << 16);                  // stride0 hi=0 | stride1 lo
  g1[7] = (int)(n8 >> 16);                              // stride1 hi
  i32x4 z4 = {0, 0, 0, 0};
  i32x8 z8 = {0, 0, 0, 0, 0, 0, 0, 0};
  __builtin_amdgcn_tensor_load_to_lds(g0, g1, z4, z4, z8, 0);
}

// ---------------------------------------------------------------------------
// Kernel 0: one-time pack of all weights to bf16, column-major, contiguous:
//   [0,8192)        Wc  (128 cols x 64 k)
//   [8192,16384)    Wn
//   [16384,32768)   Wp2 (128 x 128)
//   [32768,49152)   Wa1
//   [49152,65536)   Wa2
// Byte-identical to the LDS image the GEMMs consume -> TDM can DMA it raw.
// ---------------------------------------------------------------------------
__global__ __launch_bounds__(256)
void pack_weights_kernel(const float* __restrict__ Wc, const float* __restrict__ Wn,
                         const float* __restrict__ Wp2, const float* __restrict__ Wa1,
                         const float* __restrict__ Wa2, __bf16* __restrict__ pk) {
  int i = blockIdx.x * 256 + threadIdx.x;
  const int SZ1 = COUT * CIN;          // 8192
  const int SZ2 = COUT * COUT;         // 16384
  if (i < 2 * SZ1) {
    const float* W = (i < SZ1) ? Wc : Wn;
    int j = (i < SZ1) ? i : i - SZ1;
    int n = j >> 6, k = j & 63;
    pk[i] = (__bf16)W[k * COUT + n];
  } else {
    int j = i - 2 * SZ1;
    int sel = j / SZ2;
    j -= sel * SZ2;
    const float* W = (sel == 0) ? Wp2 : (sel == 1) ? Wa1 : Wa2;
    int n = j >> 7, k = j & 127;
    pk[i] = (__bf16)W[k * COUT + n];
  }
}

// ---------------------------------------------------------------------------
// Kernel 1: fused KNN (per-query running top-16, candidate positions tiled in LDS)
// ---------------------------------------------------------------------------
__global__ __launch_bounds__(256)
void knn_kernel(const float* __restrict__ pos, int* __restrict__ nbidx) {
  __shared__ float sp[256][4];
  int b = blockIdx.y;
  int q = blockIdx.x * 256 + threadIdx.x;
  const float* pb = pos + (size_t)b * N_PTS * 3;
  float qx = pb[q * 3 + 0], qy = pb[q * 3 + 1], qz = pb[q * 3 + 2];
  float sqq = qx * qx + qy * qy + qz * qz;
  float bestd[KNN];
  int   besti[KNN];
#pragma unroll
  for (int j = 0; j < KNN; ++j) { bestd[j] = -3.0e38f; besti[j] = 0; }
  for (int tile = 0; tile < N_PTS; tile += 256) {
    int c = tile + threadIdx.x;
    float cx = pb[c * 3 + 0], cy = pb[c * 3 + 1], cz = pb[c * 3 + 2];
    sp[threadIdx.x][0] = cx; sp[threadIdx.x][1] = cy; sp[threadIdx.x][2] = cz;
    sp[threadIdx.x][3] = cx * cx + cy * cy + cz * cz;
    __syncthreads();
    for (int j = 0; j < 256; ++j) {
      float nd = 2.f * (qx * sp[j][0] + qy * sp[j][1] + qz * sp[j][2]) - sqq - sp[j][3];
      if (nd > bestd[KNN - 1]) {              // insert (keep sorted descending)
        bestd[KNN - 1] = nd; besti[KNN - 1] = tile + j;
#pragma unroll
        for (int t = KNN - 1; t > 0; --t) {
          if (bestd[t] > bestd[t - 1]) {
            float td = bestd[t]; bestd[t] = bestd[t - 1]; bestd[t - 1] = td;
            int   ti = besti[t]; besti[t] = besti[t - 1]; besti[t - 1] = ti;
          }
        }
      }
    }
    __syncthreads();
  }
  size_t base = ((size_t)b * N_PTS + q) * KNN;
#pragma unroll
  for (int j = 0; j < KNN; ++j) nbidx[base + j] = b * N_PTS + besti[j];  // global row
}

// ---------------------------------------------------------------------------
// Kernel 2a: P1 = pos @ Wp1  (K=3, trivial scalar FMA)
// ---------------------------------------------------------------------------
__global__ __launch_bounds__(256)
void p1_kernel(const float* __restrict__ pos, const float* __restrict__ Wp1,
               float* __restrict__ P1) {
  int i = blockIdx.x * 256 + threadIdx.x;   // over R_TOT * COUT (exact grid)
  int r = i >> 7, c = i & 127;
  float px = pos[r * 3 + 0], py = pos[r * 3 + 1], pz = pos[r * 3 + 2];
  P1[i] = px * Wp1[c] + py * Wp1[COUT + c] + pz * Wp1[2 * COUT + c];
}

// ---------------------------------------------------------------------------
// Kernel 2b: feat_c = x@Wc+bc, featn = x@Wn+bn  via WMMA (bf16 in, f32 accum).
// Weights arrive in LDS via TDM (single descriptor, wave 0).
// ---------------------------------------------------------------------------
__global__ __launch_bounds__(256)
void precompute_kernel(const float* __restrict__ x,
                       const __bf16* __restrict__ pk,     // packed wc|wn (32KB)
                       const float* __restrict__ bc, const float* __restrict__ bn,
                       float* __restrict__ fc, float* __restrict__ fnb) {
  __shared__ __bf16 wpack_l[2 * COUT * CIN];   // wc col-major, then wn
  if (threadIdx.x < 32) {                      // wave 0 issues the DMA
    tdm_load_lds_1d(0u, pk, 2 * COUT * CIN * sizeof(__bf16));
    __builtin_amdgcn_s_wait_tensorcnt(0);
  }
  __syncthreads();
  const __bf16* wc_l = wpack_l;
  const __bf16* wn_l = wpack_l + COUT * CIN;

  int wave = threadIdx.x >> 5, lane = threadIdx.x & 31;
  int hi = lane >> 4, m = lane & 15;
  int row0 = (blockIdx.x * 8 + wave) * 16;
  const float* arow = x + (size_t)(row0 + m) * CIN;
  v16bf a[2];
#pragma unroll
  for (int q = 0; q < 2; ++q) {                 // A frag: lanes<16 K{0..7,16..23}
    int k0 = 32 * q + hi * 8;
#pragma unroll
    for (int i = 0; i < 8; ++i) {
      a[q][i]     = (__bf16)arow[k0 + i];
      a[q][8 + i] = (__bf16)arow[k0 + 16 + i];
    }
  }
#pragma unroll
  for (int t = 0; t < 8; ++t) {
    v8f ac = {}, an = {};
#pragma unroll
    for (int q = 0; q < 2; ++q) {
      ac = wmma_bf16(a[q], load_b_frag(wc_l, CIN, t, q, lane), ac);
      an = wmma_bf16(a[q], load_b_frag(wn_l, CIN, t, q, lane), an);
    }
    int n = 16 * t + m;
    float bcv = bc[n], bnv = bn[n];             // L2-cached scalar-ish loads
#pragma unroll
    for (int r = 0; r < 8; ++r) {               // C layout: M = r + hi*8
      size_t row = row0 + r + hi * 8;
      fc [row * COUT + n] = ac[r] + bcv;
      fnb[row * COUT + n] = an[r] + bnv;
    }
  }
}

// ---------------------------------------------------------------------------
// Kernel 3: fused per-point attention. One wave per point (M=16 neighbors).
// Three chained 16x128 @ 128x128 WMMA GEMMs (Wp2, Wa1, Wa2), softmax, aggregate.
// All three weight matrices (96KB bf16) DMA'd into LDS by TDM.
// ---------------------------------------------------------------------------
__global__ __launch_bounds__(128)
void fused_attn_kernel(const float* __restrict__ fc,  const float* __restrict__ fnb,
                       const float* __restrict__ P1,  const int* __restrict__ nbidx,
                       const __bf16* __restrict__ wpk,   // packed wp2|wa1|wa2 (96KB)
                       const float* __restrict__ bp1, const float* __restrict__ bp2,
                       const float* __restrict__ ba1, const float* __restrict__ ba2,
                       const float* __restrict__ Ws,  const float* __restrict__ bs,
                       float* __restrict__ out) {
  extern __shared__ char smem[];
  __bf16* w2   = (__bf16*)smem;                 // 3 x 128x128 bf16 col-major = 96KB
  __bf16* wa1  = w2  + COUT * COUT;
  __bf16* wa2  = wa1 + COUT * COUT;
  float*  bp1_l = (float*)(wa2 + COUT * COUT);
  float*  bp2_l = bp1_l + COUT;
  float*  ba1_l = bp2_l + COUT;
  float*  ba2_l = ba1_l + COUT;
  float*  ws_l  = ba2_l + COUT;
  char*   scr   = (char*)(ws_l + COUT);         // per-wave scratch: 8KB f32 + 4KB bf16

  int wave = threadIdx.x >> 5, lane = threadIdx.x & 31;
  int hi = lane >> 4, m = lane & 15;
  float*  stgf = (float*)(scr + (size_t)wave * (16 * COUT * 4 + 16 * COUT * 2));
  __bf16* stgh = (__bf16*)(stgf + 16 * COUT);
  float*  red  = stgf;                          // reused once stgf is dead
  float*  scs  = red + 256;

  if (threadIdx.x < 32) {                       // wave 0 DMAs all three weights
    tdm_load_lds_1d(0u, wpk, 3 * COUT * COUT * sizeof(__bf16));
    __builtin_amdgcn_s_wait_tensorcnt(0);
  }
  { int i = threadIdx.x;                        // blockDim == COUT
    bp1_l[i] = bp1[i]; bp2_l[i] = bp2[i]; ba1_l[i] = ba1[i];
    ba2_l[i] = ba2[i]; ws_l[i] = Ws[i]; }
  __syncthreads();

  int p  = blockIdx.x * 4 + wave;
  int nb = nbidx[(size_t)p * KNN + m];          // neighbor for A-frag row m

  // ---- build A = relu(P1[p] - P1[nb] + bp1) directly in fragment layout ----
  const float* p1c = P1 + (size_t)p  * COUT;
  const float* p1n = P1 + (size_t)nb * COUT;
  v16bf a[4];
#pragma unroll
  for (int q = 0; q < 4; ++q) {
    int k0 = 32 * q + hi * 8;
#pragma unroll
    for (int i = 0; i < 8; ++i) {
      float v0 = p1c[k0 + i]      - p1n[k0 + i]      + bp1_l[k0 + i];
      float v1 = p1c[k0 + 16 + i] - p1n[k0 + 16 + i] + bp1_l[k0 + 16 + i];
      a[q][i]     = (__bf16)fmaxf(v0, 0.f);
      a[q][8 + i] = (__bf16)fmaxf(v1, 0.f);
    }
  }

  // ---- GEMM1: penc = A @ Wp2 + bp2 (C layout: lane holds col n, rows r+hi*8) ----
  v8f penc[8];
#pragma unroll
  for (int t = 0; t < 8; ++t) {
    v8f acc = {};
#pragma unroll
    for (int q = 0; q < 4; ++q)
      acc = wmma_bf16(a[q], load_b_frag(w2, COUT, t, q, lane), acc);
    float bb = bp2_l[16 * t + m];
#pragma unroll
    for (int r = 0; r < 8; ++r) acc[r] += bb;
    penc[t] = acc;
  }

  // ---- gather feat_nb rows (vectorized) into stgf ----
  {
    const float* fr = fnb + (size_t)nb * COUT + hi * 64;
    float* dst = stgf + m * COUT + hi * 64;
#pragma unroll
    for (int c = 0; c < 64; c += 4)
      *(float4*)(dst + c) = *(const float4*)(fr + c);
  }
  __syncthreads();

  // ---- fnp = fnb + penc (regs); h = fc - fnb + penc -> stgh (bf16, row-major) ----
  v8f fnp[8];
#pragma unroll
  for (int t = 0; t < 8; ++t) {
    int n = 16 * t + m;
    float fcv = fc[(size_t)p * COUT + n];
#pragma unroll
    for (int r = 0; r < 8; ++r) {
      int M = r + hi * 8;
      float fv = stgf[M * COUT + n];
      float pe = penc[t][r];
      fnp[t][r] = fv + pe;
      stgh[M * COUT + n] = (__bf16)(fcv - fv + pe);
    }
  }
  __syncthreads();

  // ---- transpose-read h as A fragments ----
  const __bf16* hrow = stgh + m * COUT;
#pragma unroll
  for (int q = 0; q < 4; ++q) {
    int k0 = 32 * q + hi * 8;
#pragma unroll
    for (int i = 0; i < 8; ++i) { a[q][i] = hrow[k0 + i]; a[q][8 + i] = hrow[k0 + 16 + i]; }
  }
  __syncthreads();

  // ---- GEMM2: t1 = relu(h @ Wa1 + ba1) -> stgh ----
#pragma unroll
  for (int t = 0; t < 8; ++t) {
    v8f acc = {};
#pragma unroll
    for (int q = 0; q < 4; ++q)
      acc = wmma_bf16(a[q], load_b_frag(wa1, COUT, t, q, lane), acc);
    int n = 16 * t + m;
    float bb = ba1_l[n];
#pragma unroll
    for (int r = 0; r < 8; ++r)
      stgh[(r + hi * 8) * COUT + n] = (__bf16)fmaxf(acc[r] + bb, 0.f);
  }
  __syncthreads();
#pragma unroll
  for (int q = 0; q < 4; ++q) {
    int k0 = 32 * q + hi * 8;
#pragma unroll
    for (int i = 0; i < 8; ++i) { a[q][i] = hrow[k0 + i]; a[q][8 + i] = hrow[k0 + 16 + i]; }
  }
  __syncthreads();

  // ---- GEMM3: attn = t1 @ Wa2 + ba2; score partials ps[r] = sum_n attn*Ws[n] ----
  float ps[8];
#pragma unroll
  for (int r = 0; r < 8; ++r) ps[r] = 0.f;
#pragma unroll
  for (int t = 0; t < 8; ++t) {
    v8f acc = {};
#pragma unroll
    for (int q = 0; q < 4; ++q)
      acc = wmma_bf16(a[q], load_b_frag(wa2, COUT, t, q, lane), acc);
    int n = 16 * t + m;
    float bb = ba2_l[n], wv = ws_l[n];
#pragma unroll
    for (int r = 0; r < 8; ++r) ps[r] += (acc[r] + bb) * wv;
  }

  // ---- cross-lane score reduction ----
#pragma unroll
  for (int r = 0; r < 8; ++r) red[lane * 8 + r] = ps[r];
  __syncthreads();
  if (lane < KNN) {
    int Lb = (lane < 8) ? 0 : 16, r = lane & 7;
    float s = bs[0];
#pragma unroll
    for (int L = 0; L < 16; ++L) s += red[(Lb + L) * 8 + r];
    scs[lane] = s;
  }
  __syncthreads();

  // ---- softmax over 16 (redundant per lane) ----
  float wgt[KNN];
  float mx = -3.0e38f;
#pragma unroll
  for (int j = 0; j < KNN; ++j) mx = fmaxf(mx, scs[j]);
  float sum = 0.f;
#pragma unroll
  for (int j = 0; j < KNN; ++j) { wgt[j] = __expf(scs[j] - mx); sum += wgt[j]; }
  float inv = 1.f / sum;
  __syncthreads();

  // ---- aggregate: agg[n] = sum_M wgt[M] * fnp[M][n]; out = fc + agg ----
#pragma unroll
  for (int t = 0; t < 8; ++t) {
    float s = 0.f;
#pragma unroll
    for (int r = 0; r < 8; ++r) s += wgt[r + hi * 8] * fnp[t][r];
    red[t * 32 + lane] = s * inv;
  }
  __syncthreads();
  if (lane < 16) {
#pragma unroll
    for (int t = 0; t < 8; ++t) {
      int n = 16 * t + lane;
      float agg = red[t * 32 + lane] + red[t * 32 + 16 + lane];
      out[(size_t)p * COUT + n] = fc[(size_t)p * COUT + n] + agg;
    }
  }
}

// ---------------------------------------------------------------------------
extern "C" void kernel_launch(void* const* d_in, const int* in_sizes, int n_in,
                              void* d_out, int out_size, void* d_ws, size_t ws_size,
                              hipStream_t stream) {
  (void)in_sizes; (void)n_in; (void)out_size; (void)ws_size;
  const float* x   = (const float*)d_in[0];
  const float* pos = (const float*)d_in[1];
  const float* Wc  = (const float*)d_in[2];
  const float* bc  = (const float*)d_in[3];
  const float* Wn  = (const float*)d_in[4];
  const float* bn  = (const float*)d_in[5];
  const float* Wp1 = (const float*)d_in[6];
  const float* bp1 = (const float*)d_in[7];
  const float* Wp2 = (const float*)d_in[8];
  const float* bp2 = (const float*)d_in[9];
  const float* Wa1 = (const float*)d_in[10];
  const float* ba1 = (const float*)d_in[11];
  const float* Wa2 = (const float*)d_in[12];
  const float* ba2 = (const float*)d_in[13];
  const float* Ws  = (const float*)d_in[14];
  const float* bs  = (const float*)d_in[15];
  float* out = (float*)d_out;

  char* ws = (char*)d_ws;
  int*    nbidx = (int*)ws;    ws += (size_t)R_TOT * KNN  * sizeof(int);
  float*  fc    = (float*)ws;  ws += (size_t)R_TOT * COUT * sizeof(float);
  float*  fnb   = (float*)ws;  ws += (size_t)R_TOT * COUT * sizeof(float);
  float*  P1    = (float*)ws;  ws += (size_t)R_TOT * COUT * sizeof(float);
  __bf16* pk    = (__bf16*)ws;                   // 65536 bf16 = 128KB packed weights

  pack_weights_kernel<<<(2 * COUT * CIN + 3 * COUT * COUT) / 256, 256, 0, stream>>>(
      Wc, Wn, Wp2, Wa1, Wa2, pk);
  knn_kernel<<<dim3(N_PTS / 256, B_SZ), 256, 0, stream>>>(pos, nbidx);
  p1_kernel<<<(R_TOT * COUT) / 256, 256, 0, stream>>>(pos, Wp1, P1);
  precompute_kernel<<<R_TOT / (16 * 8), 256, 0, stream>>>(x, pk, bc, bn, fc, fnb);

  size_t smem = (size_t)3 * COUT * COUT * sizeof(__bf16)   // weights (96 KB)
              + (size_t)5 * COUT * sizeof(float)           // biases + Ws
              + (size_t)4 * (16 * COUT * sizeof(float) + 16 * COUT * sizeof(__bf16));
  fused_attn_kernel<<<R_TOT / 4, 128, smem, stream>>>(
      fc, fnb, P1, nbidx, pk + 2 * COUT * CIN,
      bp1, bp2, ba1, ba2, Ws, bs, out);
}
